// SpikingTransformerBlock_45397804319542
// MI455X (gfx1250) — 8-bit matrix operand order repaired and checked by reading, compiled, not run on hardware
//
#include <hip/hip_runtime.h>
#include <hip/hip_bf16.h>

typedef __attribute__((ext_vector_type(16))) _Float16 v16h;
typedef __attribute__((ext_vector_type(8)))  _Float16 h8;
typedef __attribute__((ext_vector_type(8)))  float    v8f;
typedef __attribute__((ext_vector_type(8)))  int      v8i;
typedef __attribute__((ext_vector_type(4)))  int      i4;
typedef __attribute__((ext_vector_type(2)))  int      i2;
typedef unsigned char u8;

constexpr int kB   = 4;
constexpr int kC   = 512;
constexpr int kT   = 4;
constexpr int kN   = 512;
constexpr int kTN  = 2048;
constexpr int kHid = 2048;
constexpr int kH   = 8;
constexpr int kD   = 64;

__global__ void pack_w_kernel(const float* __restrict__ W, _Float16* __restrict__ Wpk,
                              int K, int Cout, int total) {
  int id = blockIdx.x * blockDim.x + threadIdx.x;
  if (id >= total) return;
  int j  = id & 15;
  int L  = (id >> 4) & 31;
  int t2 = id >> 9;
  int ntiles = Cout >> 4;
  int nb = t2 % ntiles;
  int kb = t2 / ntiles;
  int col = nb * 16 + (L & 15);
  int k   = kb * 32 + j + ((j >= 8) ? 8 : 0) + ((L >= 16) ? 8 : 0);
  Wpk[id] = (_Float16)W[(size_t)col * K + k];
}

__global__ void lif_front_kernel(const float* __restrict__ in, const float* __restrict__ bnp,
                                 _Float16* __restrict__ out, int total) {
  int id = blockIdx.x * blockDim.x + threadIdx.x;
  if (id >= total) return;
  int c = id % kC;
  int n = (id / kC) % kN;
  int b = id / (kC * kN);
  float g  = bnp[c];
  float be = bnp[kC + c];
  float mu = bnp[2 * kC + c];
  float va = bnp[3 * kC + c];
  float inv = g * rsqrtf(va + 1e-5f);
  float add = be - mu * inv;
  float v = 0.0f;
  for (int t = 0; t < kT; ++t) {
    float xv = in[(((size_t)b * kC + c) * kT + t) * kN + n] * inv + add;
    v += (xv - v) * (2.0f / 3.0f);
    float s = (v >= 1.0f) ? 1.0f : 0.0f;
    out[((size_t)b * kTN + t * kN + n) * kC + c] = (_Float16)s;
    v *= (1.0f - s);
  }
}

__global__ __launch_bounds__(256) void gemm_lif_wmma(
    const _Float16* __restrict__ A, const _Float16* __restrict__ Wpk,
    const float* __restrict__ bnp, const float* __restrict__ bias,
    const float* __restrict__ res_bctn, const float* __restrict__ res_btc,
    float* __restrict__ out_f32, int out_bctn,
    const float* __restrict__ lif_bn,
    u8* __restrict__ sp_u8, u8* __restrict__ sp_u8T, _Float16* __restrict__ sp_h,
    int K, int Cout) {
  const int lane  = threadIdx.x & 31;
  const int wv    = threadIdx.x >> 5;
  const int n0    = blockIdx.x * 16;
  const int ntile = blockIdx.y * 8 + wv;
  const int b     = blockIdx.z;
  const int col   = ntile * 16 + (lane & 15);
  const int hi8   = (lane & 16) ? 8 : 0;

  float scale = 1.0f, shift = 0.0f;
  if (bnp) {
    float g = bnp[col], be = bnp[Cout + col], mu = bnp[2 * Cout + col], va = bnp[3 * Cout + col];
    scale = g * rsqrtf(va + 1e-5f);
    shift = be - mu * scale;
  }
  if (bias) shift += bias[col] * scale;
  float linv = 1.0f, ladd = 0.0f;
  if (lif_bn) {
    float g = lif_bn[col], be = lif_bn[Cout + col], mu = lif_bn[2 * Cout + col],
          va = lif_bn[3 * Cout + col];
    linv = g * rsqrtf(va + 1e-5f);
    ladd = be - mu * linv;
  }

  const int ntilesN = Cout >> 4;
  const int nkb = K >> 5;
  const _Float16* arow0 = A + ((size_t)b * kTN + n0 + (lane & 15)) * K;
  const size_t tstride = (size_t)kN * K;

  v8f acc[4];
#pragma unroll
  for (int t = 0; t < 4; ++t) acc[t] = (v8f){};

  for (int kb = 0; kb < nkb; ++kb) {
    const _Float16* bp = Wpk + (((size_t)kb * ntilesN + ntile) * 32 + lane) * 16;
    v16h bv = *(const v16h*)bp;
    if (kb + 1 < nkb)
      __builtin_prefetch(bp + (size_t)ntilesN * 32 * 16, 0, 1);
#pragma unroll
    for (int t = 0; t < 4; ++t) {
      const _Float16* ap = arow0 + (size_t)t * tstride + kb * 32 + hi8;
      v16h av;
      ((h8*)&av)[0] = *(const h8*)ap;
      ((h8*)&av)[1] = *(const h8*)(ap + 16);
      acc[t] = __builtin_amdgcn_wmma_f32_16x16x32_f16(false, av, false, bv,
                                                      (short)0, acc[t], false, false);
    }
  }

#pragma unroll
  for (int r = 0; r < 8; ++r) {
    const int nrow = n0 + r + hi8;
    float vals[4];
#pragma unroll
    for (int t = 0; t < 4; ++t) {
      float v = acc[t][r] * scale + shift;
      if (res_bctn) v += res_bctn[(((size_t)b * Cout + col) * kT + t) * kN + nrow];
      if (res_btc)  v += res_btc[((size_t)b * kTN + t * kN + nrow) * Cout + col];
      vals[t] = v;
      if (out_f32) {
        if (out_bctn) out_f32[(((size_t)b * Cout + col) * kT + t) * kN + nrow] = v;
        else          out_f32[((size_t)b * kTN + t * kN + nrow) * Cout + col] = v;
      }
    }
    if (sp_u8 || sp_u8T || sp_h) {
      float vm = 0.0f;
#pragma unroll
      for (int t = 0; t < 4; ++t) {
        float xv = vals[t] * linv + ladd;
        vm += (xv - vm) * (2.0f / 3.0f);
        float s = (vm >= 1.0f) ? 1.0f : 0.0f;
        int l = t * kN + nrow;
        if (sp_u8)  sp_u8[((size_t)b * kTN + l) * Cout + col] = (u8)s;
        if (sp_u8T) {
          int h = col >> 6, dd = col & 63;
          sp_u8T[(((size_t)b * kH + h) * kD + dd) * kTN + l] = (u8)s;
        }
        if (sp_h)   sp_h[((size_t)b * kTN + l) * Cout + col] = (_Float16)s;
        vm *= (1.0f - s);
      }
    }
  }
}

__global__ __launch_bounds__(32) void spik_attn_kernel(
    const u8* __restrict__ Q, const u8* __restrict__ Kc,
    const u8* __restrict__ Vt, _Float16* __restrict__ Os) {
  __shared__ u8 lds_a[16 * 64];
  const int lane = threadIdx.x & 31;
  const int qt = blockIdx.x, h = blockIdx.y, b = blockIdx.z;
  const int hi8 = (lane & 16) ? 8 : 0;
  const int lo = lane & 15;
  const int kboff = (lane & 16) ? 16 : 0;

  const u8* qp = Q + ((size_t)b * kTN + qt * 16 + lo) * kC + h * kD;
  v8i qa;
#pragma unroll
  for (int c4 = 0; c4 < 2; ++c4)
    ((i4*)&qa)[c4] = *(const i4*)(qp + kboff + 32 * c4);

  v8i oacc[4];
#pragma unroll
  for (int dq = 0; dq < 4; ++dq) oacc[dq] = (v8i){};
  const v8i zacc = {};

  for (int kb = 0; kb < kTN / 64; ++kb) {
#pragma unroll
    for (int st = 0; st < 4; ++st) {
      const u8* kp = Kc + ((size_t)b * kTN + kb * 64 + st * 16 + lo) * kC + h * kD;
      v8i kf;
      ((i4*)&kf)[0] = *(const i4*)(kp + kboff);
      ((i4*)&kf)[1] = *(const i4*)(kp + kboff + 32);
      v8i a32 = __builtin_amdgcn_wmma_i32_16x16x64_iu8(false, qa, false, kf,
                                                       zacc, false, false);
#pragma unroll
      for (int r = 0; r < 8; ++r)
        lds_a[(r + hi8) * 64 + st * 16 + lo] = (u8)a32[r];
    }
    __syncthreads();

    v8i a2;
#pragma unroll
    for (int c4 = 0; c4 < 4; ++c4)
      ((i2*)&a2)[c4] = *(const i2*)(&lds_a[lo * 64 + hi8 + 16 * c4]);

    const u8* vb = Vt + (((size_t)b * kH + h) * kD) * (size_t)kTN + (size_t)kb * 64;
#pragma unroll
    for (int dq = 0; dq < 4; ++dq) {
      const u8* vp = vb + (size_t)(dq * 16 + lo) * kTN;
      v8i vf;
#pragma unroll
      for (int c4 = 0; c4 < 4; ++c4)
        ((i2*)&vf)[c4] = *(const i2*)(vp + hi8 + 16 * c4);
      oacc[dq] = __builtin_amdgcn_wmma_i32_16x16x64_iu8(false, a2, false, vf,
                                                        oacc[dq], false, false);
    }
    __syncthreads();
  }

#pragma unroll
  for (int dq = 0; dq < 4; ++dq) {
#pragma unroll
    for (int r = 0; r < 8; ++r) {
      float f = (float)oacc[dq][r] * 0.125f;
      float s = (f * (2.0f / 3.0f) - 1.0f >= 0.0f) ? 1.0f : 0.0f;
      int l = qt * 16 + r + hi8;
      int c = h * kD + dq * 16 + lo;
      Os[((size_t)b * kTN + l) * kC + c] = (_Float16)s;
    }
  }
}

extern "C" void kernel_launch(void* const* d_in, const int* in_sizes, int n_in,
                              void* d_out, int out_size, void* d_ws, size_t ws_size,
                              hipStream_t stream) {
  const float* x       = (const float*)d_in[0];
  const float* attn_bn = (const float*)d_in[1];
  const float* q_w     = (const float*)d_in[2];
  const float* q_bn    = (const float*)d_in[3];
  const float* k_w     = (const float*)d_in[4];
  const float* k_bn    = (const float*)d_in[5];
  const float* v_w     = (const float*)d_in[6];
  const float* v_bn    = (const float*)d_in[7];
  const float* proj_w  = (const float*)d_in[8];
  const float* proj_b  = (const float*)d_in[9];
  const float* mlp_bn  = (const float*)d_in[10];
  const float* fc1_w   = (const float*)d_in[11];
  const float* fc1_b   = (const float*)d_in[12];
  const float* fc1_bn  = (const float*)d_in[13];
  const float* fc2_w   = (const float*)d_in[14];
  const float* fc2_b   = (const float*)d_in[15];

  char* wsb = (char*)d_ws;
  size_t off = 0;
  auto carve = [&](size_t bytes) -> char* {
    char* p = wsb + off;
    off += (bytes + 255) & ~(size_t)255;
    return p;
  };
  _Float16* wpk_q   = (_Float16*)carve((size_t)kC * kC * 2);
  _Float16* wpk_k   = (_Float16*)carve((size_t)kC * kC * 2);
  _Float16* wpk_v   = (_Float16*)carve((size_t)kC * kC * 2);
  _Float16* wpk_p   = (_Float16*)carve((size_t)kC * kC * 2);
  _Float16* wpk_fc1 = (_Float16*)carve((size_t)kC * kHid * 2);
  _Float16* wpk_fc2 = (_Float16*)carve((size_t)kHid * kC * 2);
  _Float16* xq      = (_Float16*)carve((size_t)kB * kTN * kC * 2);
  u8*       q_u8    = (u8*)      carve((size_t)kB * kTN * kC);
  u8*       k_u8    = (u8*)      carve((size_t)kB * kTN * kC);
  u8*       vT_u8   = (u8*)      carve((size_t)kB * kTN * kC);
  _Float16* o_h     = (_Float16*)carve((size_t)kB * kTN * kC * 2);
  float*    x1      = (float*)   carve((size_t)kB * kTN * kC * 4);
  _Float16* s_mlp   = (_Float16*)carve((size_t)kB * kTN * kC * 2);
  _Float16* s2      = (_Float16*)carve((size_t)kB * kTN * kHid * 2);

  const int totC = kB * kN * kC;
  const int pkCC = kC * kC;
  const int pkCH = kC * kHid;

  pack_w_kernel<<<(pkCC + 255) / 256, 256, 0, stream>>>(q_w,    wpk_q,   kC,   kC,   pkCC);
  pack_w_kernel<<<(pkCC + 255) / 256, 256, 0, stream>>>(k_w,    wpk_k,   kC,   kC,   pkCC);
  pack_w_kernel<<<(pkCC + 255) / 256, 256, 0, stream>>>(v_w,    wpk_v,   kC,   kC,   pkCC);
  pack_w_kernel<<<(pkCC + 255) / 256, 256, 0, stream>>>(proj_w, wpk_p,   kC,   kC,   pkCC);
  pack_w_kernel<<<(pkCH + 255) / 256, 256, 0, stream>>>(fc1_w,  wpk_fc1, kC,   kHid, pkCH);
  pack_w_kernel<<<(pkCH + 255) / 256, 256, 0, stream>>>(fc2_w,  wpk_fc2, kHid, kC,   pkCH);

  lif_front_kernel<<<(totC + 255) / 256, 256, 0, stream>>>(x, attn_bn, xq, totC);

  dim3 gC(kN / 16, kC / 128, kB);
  gemm_lif_wmma<<<gC, 256, 0, stream>>>(xq, wpk_q, q_bn, nullptr, nullptr, nullptr,
                                        nullptr, 0, nullptr, q_u8, nullptr, nullptr, kC, kC);
  gemm_lif_wmma<<<gC, 256, 0, stream>>>(xq, wpk_k, k_bn, nullptr, nullptr, nullptr,
                                        nullptr, 0, nullptr, k_u8, nullptr, nullptr, kC, kC);
  gemm_lif_wmma<<<gC, 256, 0, stream>>>(xq, wpk_v, v_bn, nullptr, nullptr, nullptr,
                                        nullptr, 0, nullptr, nullptr, vT_u8, nullptr, kC, kC);

  spik_attn_kernel<<<dim3(kTN / 16, kH, kB), 32, 0, stream>>>(q_u8, k_u8, vT_u8, o_h);

  gemm_lif_wmma<<<gC, 256, 0, stream>>>(o_h, wpk_p, nullptr, proj_b, x, nullptr,
                                        x1, 0, mlp_bn, nullptr, nullptr, s_mlp, kC, kC);

  dim3 gH(kN / 16, kHid / 128, kB);
  gemm_lif_wmma<<<gH, 256, 0, stream>>>(s_mlp, wpk_fc1, fc1_bn, fc1_b, nullptr, nullptr,
                                        nullptr, 0, nullptr, nullptr, nullptr, s2, kC, kHid);

  gemm_lif_wmma<<<gC, 256, 0, stream>>>(s2, wpk_fc2, nullptr, fc2_b, nullptr, x1,
                                        (float*)d_out, 1, nullptr, nullptr, nullptr, nullptr,
                                        kHid, kC);
}
